// DiscRNNG_35974646071516
// MI455X (gfx1250) — matrix operand order repaired and checked by reading, not compiled, not run on hardware
//
#include <hip/hip_runtime.h>

// ---------------------------------------------------------------------------
// DiscRNNG forward for MI455X (gfx1250, wave32, WMMA).
// Phase A (parallel, WMMA bf16):  embeddings -> xw/xa -> per-LSTM input gates GX
// Phase B (sequential, LDS-resident weights): 3 independent LSTM recurrences,
//          each split over 16 WGPs with per-step atomic grid barrier.
// Phase C (parallel, WMMA bf16):  summary GEMM (tanh) -> logits GEMM -> log_softmax
// ---------------------------------------------------------------------------

#define TSTEPS 4096
#define HDIM   512
#define NACT   100

typedef __attribute__((ext_vector_type(16))) __bf16 v16bf;
typedef __attribute__((ext_vector_type(8)))  float  v8f;

static __device__ __forceinline__ unsigned short f2bf(float x) {
  unsigned u = __float_as_uint(x);
  u += 0x7FFFu + ((u >> 16) & 1u);          // round-to-nearest-even
  return (unsigned short)(u >> 16);
}

union FragBF { v16bf v; uint4 q[2]; };

// ---------------------------------------------------------------------------
// Generic GEMM:  C[M,N] = act(A[M,K]_bf16 @ W[N,K]_bf16^T + b1[n] + b2[n])
// Block = 256 threads = 8 waves; block tile 128x128; wave tile 32x64.
// ACT: 0=none 1=relu 2=tanh.  M,N multiples of 128; K multiple of 32.
// ---------------------------------------------------------------------------
template<int ACT, bool OUTBF16>
__global__ void __launch_bounds__(256)
gemm_bf16(const unsigned short* __restrict__ A, const unsigned short* __restrict__ W,
          const float* __restrict__ b1, const float* __restrict__ b2,
          void* __restrict__ Cout, int K, int lda, int ldc)
{
  const int lane = threadIdx.x & 31;
  const int wid  = threadIdx.x >> 5;
  const int wm = wid & 3;                  // 4 waves along M
  const int wn = wid >> 2;                 // 2 waves along N
  const int row0 = blockIdx.x * 128 + wm * 32;
  const int col0 = blockIdx.y * 128 + wn * 64;

  v8f acc[2][4];
  #pragma unroll
  for (int i = 0; i < 2; ++i)
    #pragma unroll
    for (int j = 0; j < 4; ++j) acc[i][j] = (v8f)0.f;

  const int arow = lane & 15;
  const int aoff = (lane < 16) ? 0 : 8;    // ISA A 16x32 bf16 per-lane layout
  const int bklo = (lane < 16) ? 0 : 8;    // ISA B 32x16 bf16 per-lane layout
  const int bcol = lane & 15;

  for (int kb = 0; kb < K; kb += 32) {
    FragBF a[2];
    #pragma unroll
    for (int i = 0; i < 2; ++i) {
      const unsigned short* ap = A + (size_t)(row0 + 16 * i + arow) * lda + kb + aoff;
      a[i].q[0] = *(const uint4*)(ap);        // K = off..off+7
      a[i].q[1] = *(const uint4*)(ap + 16);   // K = off+16..off+23
    }
    FragBF b[4];
    #pragma unroll
    for (int j = 0; j < 4; ++j) {
      const unsigned short* bp = W + (size_t)(col0 + 16 * j + bcol) * K + kb + bklo;
      b[j].q[0] = *(const uint4*)(bp);        // K = klo..klo+7
      b[j].q[1] = *(const uint4*)(bp + 16);   // K = klo+16..klo+23
    }
    #pragma unroll
    for (int i = 0; i < 2; ++i)
      #pragma unroll
      for (int j = 0; j < 4; ++j)
        acc[i][j] = __builtin_amdgcn_wmma_f32_16x16x32_bf16(
            false, a[i].v, false, b[j].v, (short)0, acc[i][j], false, false);
  }

  #pragma unroll
  for (int j = 0; j < 4; ++j) {
    const int ccol = col0 + 16 * j + (lane & 15);
    float bias = (b1 ? b1[ccol] : 0.f) + (b2 ? b2[ccol] : 0.f);
    #pragma unroll
    for (int i = 0; i < 2; ++i) {
      #pragma unroll
      for (int r = 0; r < 8; ++r) {
        const int crow = row0 + 16 * i + ((lane < 16) ? r : 8 + r);
        float v = acc[i][j][r] + bias;
        if (ACT == 1) v = fmaxf(v, 0.f);
        else if (ACT == 2) v = tanhf(v);
        if (OUTBF16) ((unsigned short*)Cout)[(size_t)crow * ldc + ccol] = f2bf(v);
        else         ((float*)Cout)[(size_t)crow * ldc + ccol] = v;
      }
    }
  }
}

// ---------------------------------------------------------------------------
// fp32 -> bf16 weight conversion with zero padding to [Np,Kp]
// ---------------------------------------------------------------------------
__global__ void cvt_pad_bf16(unsigned short* __restrict__ dst, const float* __restrict__ src,
                             int N, int K, int Np, int Kp)
{
  const size_t total = (size_t)Np * Kp;
  for (size_t i = (size_t)blockIdx.x * blockDim.x + threadIdx.x; i < total;
       i += (size_t)gridDim.x * blockDim.x) {
    int n = (int)(i / Kp), k = (int)(i % Kp);
    float v = (n < N && k < K) ? src[(size_t)n * K + k] : 0.f;
    dst[i] = f2bf(v);
  }
}

__global__ void pad_bias_f32(float* dst, const float* src, int N, int Np)
{
  int i = blockIdx.x * blockDim.x + threadIdx.x;
  if (i < Np) dst[i] = (i < N) ? src[i] : 0.f;
}

// ---------------------------------------------------------------------------
// Embedding gather -> bf16 A matrices (Ew: [T,352] with pad, Ea: [T,64])
// ---------------------------------------------------------------------------
__global__ void gather_embed(const int* __restrict__ words, const int* __restrict__ pos,
                             const int* __restrict__ acts,
                             const float* __restrict__ wemb, const float* __restrict__ pemb,
                             const float* __restrict__ aemb,
                             unsigned short* __restrict__ Ew, unsigned short* __restrict__ Ea)
{
  const int t = blockIdx.x;
  const int w = words[t], p = pos[t], a = acts[t];
  for (int c = threadIdx.x; c < 352; c += blockDim.x) {
    float v;
    if (c < 300)      v = wemb[(size_t)w * 300 + c];
    else if (c < 332) v = pemb[p * 32 + (c - 300)];
    else              v = 0.f;
    Ew[(size_t)t * 352 + c] = f2bf(v);
  }
  for (int c = threadIdx.x; c < 64; c += blockDim.x)
    Ea[(size_t)t * 64 + c] = f2bf(aemb[a * 64 + c]);
}

// ---------------------------------------------------------------------------
// Sequential phase. 3 LSTMs x 16 WGs; each WG owns 32 hidden units =>
// 128 whh rows (fp32, 256KB) resident in LDS for all 4096 steps.
// Per step: matvec slice from LDS, gate nonlinearity, write h slice,
// atomic grid barrier among the 16 WGs of this LSTM.
// ---------------------------------------------------------------------------
#define GSEQ 16
#define SEQT 256
#define RSTR 520   // LDS row stride in floats (512 + 8 pad -> no bank conflicts)

struct SeqParams {
  const float* whh[3]; const float* bih[3]; const float* bhh[3];
  const float* h0[3];  const float* c0[3];  const float* gx[3];
  float* hall[3]; float* hbuf[3];
  unsigned* cnt; unsigned* phase;
};

static __device__ __forceinline__ void grid_barrier(unsigned* cnt, unsigned* phase,
                                                    unsigned want, int tid)
{
  __syncthreads();
  if (tid == 0) {
    unsigned old = __hip_atomic_fetch_add(cnt, 1u, __ATOMIC_ACQ_REL, __HIP_MEMORY_SCOPE_AGENT);
    if (old == GSEQ - 1u) {
      __hip_atomic_store(cnt, 0u, __ATOMIC_RELAXED, __HIP_MEMORY_SCOPE_AGENT);
      __hip_atomic_fetch_add(phase, 1u, __ATOMIC_ACQ_REL, __HIP_MEMORY_SCOPE_AGENT);
    }
    while (__hip_atomic_load(phase, __ATOMIC_ACQUIRE, __HIP_MEMORY_SCOPE_AGENT) < want)
      __builtin_amdgcn_s_sleep(1);
  }
  __syncthreads();
}

__global__ void __launch_bounds__(SEQT) lstm_seq(SeqParams P)
{
  const int l = blockIdx.x / GSEQ;     // which LSTM
  const int g = blockIdx.x % GSEQ;     // which hidden slice
  const int tid = threadIdx.x;

  extern __shared__ float smem[];
  float* wsl  = smem;                  // [128][RSTR]
  float* h_sh = wsl + 128 * RSTR;      // [512]
  float* gv   = h_sh + 512;            // [128] gate pre-activations
  float* gxb  = gv + 128;              // [128] bih+bhh slice

  const float* whh = P.whh[l];
  const float* gx  = P.gx[l];
  float* hall = P.hall[l];
  float* hbuf = P.hbuf[l];
  unsigned* cnt = P.cnt + l;
  unsigned* phs = P.phase + l;

  // Preload weight slice: LDS row r = gate*32 + jj -> whh row R = gate*512 + g*32 + jj
  for (int i = tid; i < 128 * 512; i += SEQT) {
    int r = i >> 9, k = i & 511;
    int R = ((r >> 5) << 9) + (g << 5) + (r & 31);
    wsl[r * RSTR + k] = whh[(size_t)R * 512 + k];
  }
  if (tid < 128) {
    int R = ((tid >> 5) << 9) + (g << 5) + (tid & 31);
    gxb[tid] = P.bih[l][R] + P.bhh[l][R];
  }
  float c_reg = 0.f;
  if (tid < 32) {
    int j = (g << 5) + tid;
    c_reg = P.c0[l][j];
    float h0v = P.h0[l][j];
    hbuf[j] = h0v;                         // parity-0 buffer
    hall[j] = h0v;                         // H state index 0 (= h before step 0)
  }
  __threadfence();
  grid_barrier(cnt, phs, 1u, tid);

  const int r = tid >> 1, half = tid & 1;  // 2 threads per gate row
  const int Rr = ((r >> 5) << 9) + (g << 5) + (r & 31);
  const float* wr = wsl + r * RSTR + half;

  for (int t = 0; t < TSTEPS; ++t) {
    const float* hsrc = hbuf + ((t & 1) << 9);
    for (int i = tid; i < 512; i += SEQT) h_sh[i] = hsrc[i];
    __syncthreads();

    float s = 0.f;
    #pragma unroll 8
    for (int j2 = 0; j2 < 256; ++j2)
      s += wr[2 * j2] * h_sh[half + 2 * j2];   // interleaved halves: bank-conflict free
    s += __shfl_xor(s, 1, 32);
    if (!half) gv[r] = s + gx[(size_t)t * 2048 + Rr] + gxb[r];
    __syncthreads();

    if (tid < 32) {
      float iv = gv[tid], fv = gv[32 + tid], gg = gv[64 + tid], ov = gv[96 + tid];
      float ig = 1.f / (1.f + __expf(-iv));
      float fg = 1.f / (1.f + __expf(-fv));
      float og = 1.f / (1.f + __expf(-ov));
      c_reg = fg * c_reg + ig * tanhf(gg);
      float h2 = og * tanhf(c_reg);
      int j = (g << 5) + tid;
      hbuf[(((t + 1) & 1) << 9) + j] = h2;
      hall[(size_t)(t + 1) * 512 + j] = h2;
    }
    __threadfence();
    grid_barrier(cnt, phs, (unsigned)(t + 2), tid);
  }
}

// ---------------------------------------------------------------------------
// Build TOP[t] = bf16(concat(Hs[t], Hb[t], Hh[t])), t uses PRE-step states.
// ---------------------------------------------------------------------------
__global__ void build_top(const float* __restrict__ hs, const float* __restrict__ hb,
                          const float* __restrict__ hh, unsigned short* __restrict__ top)
{
  const size_t total = (size_t)TSTEPS * 1536;
  for (size_t i = (size_t)blockIdx.x * blockDim.x + threadIdx.x; i < total;
       i += (size_t)gridDim.x * blockDim.x) {
    int t = (int)(i / 1536), c = (int)(i % 1536);
    const float* src = (c < 512) ? hs : (c < 1024 ? hb : hh);
    top[i] = f2bf(src[(size_t)t * 512 + (c & 511)]);
  }
}

// ---------------------------------------------------------------------------
// Row-wise log_softmax over 100 logits (one wave per row, shuffle reductions)
// ---------------------------------------------------------------------------
__global__ void __launch_bounds__(256) logsoftmax_k(const float* __restrict__ logits,
                                                    float* __restrict__ out, int ldl)
{
  const int lane = threadIdx.x & 31;
  const int wid  = threadIdx.x >> 5;
  const int row  = blockIdx.x * 8 + wid;
  if (row >= TSTEPS) return;
  const float* x = logits + (size_t)row * ldl;
  float xv[4]; float m = -1e30f;
  #pragma unroll
  for (int j = 0; j < 4; ++j) {
    int c = lane + 32 * j;
    xv[j] = (c < NACT) ? x[c] : -1e30f;
    m = fmaxf(m, xv[j]);
  }
  for (int o = 16; o; o >>= 1) m = fmaxf(m, __shfl_xor(m, o, 32));
  float s = 0.f;
  #pragma unroll
  for (int j = 0; j < 4; ++j) { int c = lane + 32 * j; if (c < NACT) s += __expf(xv[j] - m); }
  for (int o = 16; o; o >>= 1) s += __shfl_xor(s, o, 32);
  float lse = m + __logf(s);
  #pragma unroll
  for (int j = 0; j < 4; ++j) {
    int c = lane + 32 * j;
    if (c < NACT) out[(size_t)row * NACT + c] = xv[j] - lse;
  }
}

// ---------------------------------------------------------------------------
extern "C" void kernel_launch(void* const* d_in, const int* in_sizes, int n_in,
                              void* d_out, int out_size, void* d_ws, size_t ws_size,
                              hipStream_t stream)
{
  (void)in_sizes; (void)n_in; (void)out_size;
  const int* words = (const int*)d_in[0];
  const int* pos   = (const int*)d_in[1];
  const int* acts  = (const int*)d_in[2];
  const float* wemb = (const float*)d_in[3];
  const float* pemb = (const float*)d_in[4];
  const float* aemb = (const float*)d_in[5];
  const float* w2e_w = (const float*)d_in[6];
  const float* w2e_b = (const float*)d_in[7];
  const float* a2e_w = (const float*)d_in[8];
  const float* a2e_b = (const float*)d_in[9];
  const float* wih[3] = { (const float*)d_in[10], (const float*)d_in[16], (const float*)d_in[22] };
  const float* whh[3] = { (const float*)d_in[11], (const float*)d_in[17], (const float*)d_in[23] };
  const float* bih[3] = { (const float*)d_in[12], (const float*)d_in[18], (const float*)d_in[24] };
  const float* bhh[3] = { (const float*)d_in[13], (const float*)d_in[19], (const float*)d_in[25] };
  const float* h0[3]  = { (const float*)d_in[14], (const float*)d_in[20], (const float*)d_in[26] };
  const float* c0[3]  = { (const float*)d_in[15], (const float*)d_in[21], (const float*)d_in[27] };
  const float* sum_w = (const float*)d_in[28];
  const float* sum_b = (const float*)d_in[29];
  const float* out_w = (const float*)d_in[30];
  const float* out_b = (const float*)d_in[31];

  // ---- workspace carve-up (256B aligned) ----
  char* ws = (char*)d_ws; size_t off = 0;
  auto take = [&](size_t bytes) -> char* {
    char* p = ws + off; off = (off + bytes + 255) & ~(size_t)255; return p;
  };
  unsigned* sync      = (unsigned*)take(64);                               // cnt[3] @0, phase[3] @ +8
  unsigned short* Ew  = (unsigned short*)take((size_t)TSTEPS * 352 * 2);
  unsigned short* Ea  = (unsigned short*)take((size_t)TSTEPS * 64 * 2);
  unsigned short* w2e_bf = (unsigned short*)take((size_t)512 * 352 * 2);
  unsigned short* a2e_bf = (unsigned short*)take((size_t)512 * 64 * 2);
  unsigned short* XW  = (unsigned short*)take((size_t)TSTEPS * 512 * 2);
  unsigned short* XA  = (unsigned short*)take((size_t)TSTEPS * 512 * 2);
  unsigned short* wih_bf[3]; float* GX[3]; float* Hall[3]; float* Hbuf[3];
  for (int l = 0; l < 3; ++l) wih_bf[l] = (unsigned short*)take((size_t)2048 * 512 * 2);
  for (int l = 0; l < 3; ++l) GX[l]   = (float*)take((size_t)TSTEPS * 2048 * 4);
  for (int l = 0; l < 3; ++l) Hall[l] = (float*)take((size_t)(TSTEPS + 1) * 512 * 4);
  for (int l = 0; l < 3; ++l) Hbuf[l] = (float*)take((size_t)2 * 512 * 4);
  unsigned short* sumw_bf = (unsigned short*)take((size_t)512 * 1536 * 2);
  unsigned short* TOP = (unsigned short*)take((size_t)TSTEPS * 1536 * 2);
  unsigned short* Sbf = (unsigned short*)take((size_t)TSTEPS * 512 * 2);
  unsigned short* outw_bf = (unsigned short*)take((size_t)128 * 512 * 2);
  float* outb_pad = (float*)take(128 * 4);
  float* LOGITS   = (float*)take((size_t)TSTEPS * 128 * 4);
  if (off > ws_size) return;   // workspace too small: bail deterministically

  hipMemsetAsync(sync, 0, 64, stream);   // barrier counters start at 0 every launch

  // ---- weight conversion (fp32 -> bf16, zero-padded) ----
  auto cvt = [&](unsigned short* dst, const float* src, int N, int K, int Np, int Kp) {
    size_t total = (size_t)Np * Kp;
    int blocks = (int)((total + 255) / 256); if (blocks > 4096) blocks = 4096;
    cvt_pad_bf16<<<blocks, 256, 0, stream>>>(dst, src, N, K, Np, Kp);
  };
  cvt(w2e_bf, w2e_w, 512, 332, 512, 352);
  cvt(a2e_bf, a2e_w, 512, 64, 512, 64);
  for (int l = 0; l < 3; ++l) cvt(wih_bf[l], wih[l], 2048, 512, 2048, 512);
  cvt(sumw_bf, sum_w, 512, 1536, 512, 1536);
  cvt(outw_bf, out_w, 100, 512, 128, 512);
  pad_bias_f32<<<1, 128, 0, stream>>>(outb_pad, out_b, 100, 128);

  // ---- embedding gather ----
  gather_embed<<<TSTEPS, 128, 0, stream>>>(words, pos, acts, wemb, pemb, aemb, Ew, Ea);

  // ---- Phase A GEMMs (WMMA bf16) ----
  gemm_bf16<1, true><<<dim3(TSTEPS / 128, 512 / 128), 256, 0, stream>>>(
      Ew, w2e_bf, w2e_b, nullptr, XW, 352, 352, 512);            // xw = relu(...)
  gemm_bf16<1, true><<<dim3(TSTEPS / 128, 512 / 128), 256, 0, stream>>>(
      Ea, a2e_bf, a2e_b, nullptr, XA, 64, 64, 512);              // xa = relu(...)
  const unsigned short* gxA[3] = { XW, XW, XA };
  for (int l = 0; l < 3; ++l)
    gemm_bf16<0, false><<<dim3(TSTEPS / 128, 2048 / 128), 256, 0, stream>>>(
        gxA[l], wih_bf[l], bih[l], bhh[l], GX[l], 512, 512, 2048); // x@wih.T + b

  // ---- Phase B: sequential recurrences (LDS-resident whh) ----
  SeqParams P;
  for (int l = 0; l < 3; ++l) {
    P.whh[l] = whh[l]; P.bih[l] = bih[l]; P.bhh[l] = bhh[l];
    P.h0[l] = h0[l]; P.c0[l] = c0[l]; P.gx[l] = GX[l];
    P.hall[l] = Hall[l]; P.hbuf[l] = Hbuf[l];
  }
  P.cnt = sync; P.phase = sync + 8;
  const size_t smem_bytes = (size_t)(128 * RSTR + 512 + 128 + 128) * sizeof(float); // ~263KB <= 320KB/WGP
  hipFuncSetAttribute((const void*)lstm_seq, hipFuncAttributeMaxDynamicSharedMemorySize,
                      (int)smem_bytes);
  lstm_seq<<<3 * GSEQ, SEQT, smem_bytes, stream>>>(P);

  // ---- Phase C: summary / logits / log_softmax ----
  build_top<<<4096, 256, 0, stream>>>(Hall[0], Hall[1], Hall[2], TOP);
  gemm_bf16<2, true><<<dim3(TSTEPS / 128, 512 / 128), 256, 0, stream>>>(
      TOP, sumw_bf, sum_b, nullptr, Sbf, 1536, 1536, 512);       // tanh(top@sum_w.T+b)
  gemm_bf16<0, false><<<dim3(TSTEPS / 128, 1), 256, 0, stream>>>(
      Sbf, outw_bf, outb_pad, nullptr, LOGITS, 512, 512, 128);   // logits (N padded 128)
  logsoftmax_k<<<TSTEPS / 8, 256, 0, stream>>>(LOGITS, (float*)d_out, 128);
}
